// VelocityGrid_11158325035429
// MI455X (gfx1250) — matrix operand order repaired and checked by reading, not compiled, not run on hardware
//
#include <hip/hip_runtime.h>
#include <hip/hip_bf16.h>

typedef __attribute__((ext_vector_type(16))) _Float16 v16h;
typedef __attribute__((ext_vector_type(8)))  _Float16 v8h;
typedef __attribute__((ext_vector_type(8)))  float    v8f;
typedef __attribute__((ext_vector_type(4)))  int      v4i;

typedef __attribute__((address_space(1))) v4i* as1_b128_t;
typedef __attribute__((address_space(3))) v4i* as3_b128_t;

#define GRID_DIM    512            // GX == GZ == 512
#define NPOINT      131072
#define WG_THREADS  256            // 8 wave32
#define PTS_PER_WG  128            // 16 points per wave
#define WSTRIDE     520            // 512 halves + 8 pad: 16B align kept, banks rotated
#define BSTRIDE     520            // same padding for the B slab
#define INV_SMOOTH  5.0f           // 1/0.2

#if defined(__gfx1250__) && __has_builtin(__builtin_amdgcn_global_load_async_to_lds_b128)
#define ASYNC_LDS 1
#else
#define ASYNC_LDS 0
#endif

__device__ __forceinline__ float sigmoidf_fast(float a) {
    // exp(+big)->inf => 0 ; exp(-big)->0 => 1 : matches reference saturation
    return 1.0f / (1.0f + __expf(-a));
}

__device__ __forceinline__ void wait_async_lds() {
#if ASYNC_LDS
#if __has_builtin(__builtin_amdgcn_s_wait_asynccnt)
    __builtin_amdgcn_s_wait_asynccnt(0);
#else
    asm volatile("s_wait_asynccnt 0" ::: "memory");
#endif
#endif
}

// Stage one 16-column slab of Vt (16 rows x 512 halves) into LDS, all 256
// threads cooperating: 1024 chunks of 16B, 4 per thread, fully coalesced.
__device__ __forceinline__ void slab_fill(const _Float16* __restrict__ Vt,
                                          _Float16* slab, int jt, int t) {
#pragma unroll
    for (int i = 0; i < 4; ++i) {
        const int c   = i * WG_THREADS + t;
        const int row = c >> 6;              // 0..15
        const int off = (c & 63) * 8;        // halves within the 1KB row
        const _Float16* g = Vt + (jt * 16 + row) * GRID_DIM + off;
        _Float16*       l = slab + row * BSTRIDE + off;
#if ASYNC_LDS
        __builtin_amdgcn_global_load_async_to_lds_b128(
            (as1_b128_t)(v4i*)g,       // global source (AS1, b128-typed)
            (as3_b128_t)(v4i*)l,       // LDS destination (AS3, b128-typed)
            0, 0);
#else
        *(v8h*)l = *(const v8h*)g;           // global_load_b128 + ds_store_b128
#endif
    }
}

// ---------------------------------------------------------------------------
// Kernel 1: V (f32, [z][x]) -> Vt (f16, [x][z]) so a WMMA B-fragment lane
// (fixed column n, K along z) is two contiguous 16B runs.
// ---------------------------------------------------------------------------
__global__ void vt_f16_transpose_kernel(const float* __restrict__ V,
                                        _Float16* __restrict__ Vt) {
    int x = blockIdx.x * 16 + threadIdx.x;
    int z = blockIdx.y * 16 + threadIdx.y;
    Vt[x * GRID_DIM + z] = (_Float16)V[z * GRID_DIM + x];
}

// ---------------------------------------------------------------------------
// Kernel 2: fused soft-bin weights + bilinear form via v_wmma_f32_16x16x32_f16
// with async-LDS double-buffered staging of the V slabs.
// ---------------------------------------------------------------------------
__global__ void __launch_bounds__(WG_THREADS)
velocity_grid_wmma_kernel(const float* __restrict__ point,    // [N][2] (x,z)
                          const float* __restrict__ xmodel,   // [513]
                          const float* __restrict__ zmodel,   // [513]
                          const _Float16* __restrict__ Vt,    // [x][z] f16
                          float* __restrict__ out) {          // [N]
    __shared__ _Float16 zsh[PTS_PER_WG * WSTRIDE];   // z-weights (WMMA A)
    __shared__ _Float16 xsh[PTS_PER_WG * WSTRIDE];   // x-weights (final reduce)
    __shared__ _Float16 bslab[2][16 * BSTRIDE];      // double-buffered V slab
    __shared__ float    xedge[513];
    __shared__ float    zedge[513];

    const int t = threadIdx.x;               // 0..255

    for (int i = t; i < 513; i += WG_THREADS) {
        xedge[i] = xmodel[i];
        zedge[i] = zmodel[i];
    }
    __syncthreads();

    // ---- Phase 1: one (point, axis) per thread: 128 pts x 2 axes = 256 ----
    {
        const int p  = t >> 1;
        const int ax = t & 1;                // 0 -> x, 1 -> z
        const int gp = blockIdx.x * PTS_PER_WG + p;
        const float c = point[gp * 2 + ax];
        const float* eg = ax ? zedge : xedge;
        _Float16* row  = (ax ? zsh : xsh) + p * WSTRIDE;

        // pass 1: normalizer in f32 (boundary sums ~e^-50 underflow f16)
        float sum = 0.0f;
        float sp = sigmoidf_fast((c - eg[0]) * INV_SMOOTH);
        for (int i = 0; i < GRID_DIM; ++i) {
            float sn = sigmoidf_fast((c - eg[i + 1]) * INV_SMOOTH);
            sum += sp * (1.0f - sn);
            sp = sn;
        }
        const float inv = 1.0f / sum;

        // pass 2: normalized f16 weights to LDS
        sp = sigmoidf_fast((c - eg[0]) * INV_SMOOTH);
        for (int i = 0; i < GRID_DIM; ++i) {
            float sn = sigmoidf_fast((c - eg[i + 1]) * INV_SMOOTH);
            row[i] = (_Float16)(sp * (1.0f - sn) * inv);
            sp = sn;
        }
    }
    __syncthreads();

    // Kick off the first slab while we assemble A fragments.
    slab_fill(Vt, bslab[0], 0, t);

    // ---- Phase 2: per-wave 16x512 @ 512x512, f32 accumulate ----
    const int lane = t & 31;
    const int wave = t >> 5;                 // 0..7
    const int m16  = lane & 15;              // A row / C column / B column
    const bool lo  = lane < 16;

    // A fragments (16-bit A 16x32 layout):
    // lanes 0-15: e0..7 -> K 0..7,  e8..15 -> K 16..23
    // lanes16-31: e0..7 -> K 8..15, e8..15 -> K 24..31
    const _Float16* zrow = zsh + (wave * 16 + m16) * WSTRIDE;
    v16h A[16];
#pragma unroll
    for (int kt = 0; kt < 16; ++kt) {
        v8h a0 = *(const v8h*)(zrow + kt * 32 + (lo ? 0 : 8));
        v8h a1 = *(const v8h*)(zrow + kt * 32 + (lo ? 16 : 24));
        A[kt] = __builtin_shufflevector(a0, a1,
                    0, 1, 2, 3, 4, 5, 6, 7, 8, 9, 10, 11, 12, 13, 14, 15);
    }

    float acc[8] = {0.f, 0.f, 0.f, 0.f, 0.f, 0.f, 0.f, 0.f};
    int cur = 0;

    for (int jt = 0; jt < GRID_DIM / 16; ++jt) {
        wait_async_lds();       // our pending fills of bslab[cur] are done
        __syncthreads();        // everyone's fills done; everyone done reading bslab[cur^1]
        if (jt + 1 < GRID_DIM / 16)
            slab_fill(Vt, bslab[cur ^ 1], jt + 1, t);   // prefetch next slab

        v8f cfrag = {};
#pragma unroll
        for (int kt = 0; kt < 16; ++kt) {
            // B 32x16 fragment: lane = column n (= m16), same K order as A:
            // lanes 0-15 K 0-7,16-23; lanes 16-31 K 8-15,24-31.
            const _Float16* bp = &bslab[cur][m16 * BSTRIDE + kt * 32];
            v8h b0 = *(const v8h*)(bp + (lo ? 0 : 8));
            v8h b1 = *(const v8h*)(bp + (lo ? 16 : 24));
            v16h b = __builtin_shufflevector(b0, b1,
                        0, 1, 2, 3, 4, 5, 6, 7, 8, 9, 10, 11, 12, 13, 14, 15);
            cfrag = __builtin_amdgcn_wmma_f32_16x16x32_f16(
                        false, A[kt], false, b, (short)0, cfrag, false, false);
        }
        // C/D layout: VGPR r -> M = r (lanes 0-15) / r+8 (lanes 16-31), N = m16
#pragma unroll
        for (int r = 0; r < 8; ++r) {
            const int m = r + (lo ? 0 : 8);
            float xw = (float)xsh[(wave * 16 + m) * WSTRIDE + jt * 16 + m16];
            acc[r] += cfrag[r] * xw;
        }
        cur ^= 1;
    }

    // Reduce across the 16 N-lanes of each half-wave, one value per point.
#pragma unroll
    for (int r = 0; r < 8; ++r) {
        float v = acc[r];
        v += __shfl_xor(v, 1, 16);
        v += __shfl_xor(v, 2, 16);
        v += __shfl_xor(v, 4, 16);
        v += __shfl_xor(v, 8, 16);
        if (m16 == 0) {
            const int m = r + (lo ? 0 : 8);
            out[blockIdx.x * PTS_PER_WG + wave * 16 + m] = v;
        }
    }
}

extern "C" void kernel_launch(void* const* d_in, const int* in_sizes, int n_in,
                              void* d_out, int out_size, void* d_ws, size_t ws_size,
                              hipStream_t stream) {
    const float* point = (const float*)d_in[0];   // [131072, 2]
    const float* vel   = (const float*)d_in[1];   // [512, 512]
    const float* xm    = (const float*)d_in[2];   // [513]
    const float* zm    = (const float*)d_in[3];   // [513]
    float*       out   = (float*)d_out;           // [131072]
    _Float16*    Vt    = (_Float16*)d_ws;         // 512*512*2 = 512 KB scratch

    dim3 tb(16, 16), tg(GRID_DIM / 16, GRID_DIM / 16);
    vt_f16_transpose_kernel<<<tg, tb, 0, stream>>>(vel, Vt);

    velocity_grid_wmma_kernel<<<NPOINT / PTS_PER_WG, WG_THREADS, 0, stream>>>(
        point, xm, zm, Vt, out);
}
